// OursAttention_34119220199803
// MI455X (gfx1250) — matrix operand order repaired and checked by reading, not compiled, not run on hardware
//
#include <hip/hip_runtime.h>

#define B_  8
#define T_  4096
#define C_  1024
#define K_  2048   // T - requested_r (requested_r = 2048 in setup)

typedef __attribute__((ext_vector_type(16))) _Float16 v16h;
typedef __attribute__((ext_vector_type(8)))  _Float16 v8h;
typedef __attribute__((ext_vector_type(8)))  float    v8f;

// ---------------------------------------------------------------------------
// K1: per-token squared L2 norm (scores) + reciprocal norm. One wave per token.
__global__ __launch_bounds__(256) void k_scores(const float* __restrict__ x,
                                                float* __restrict__ scores,
                                                float* __restrict__ rnorm) {
    int token = blockIdx.x * 8 + (threadIdx.x >> 5);   // B_*T_/8 blocks
    int lane  = threadIdx.x & 31;
    const float4* px = (const float4*)(x + (size_t)token * C_);
    float ss = 0.f;
#pragma unroll
    for (int i = 0; i < 8; ++i) {
        float4 v = px[lane + i * 32];
        ss += v.x * v.x + v.y * v.y + v.z * v.z + v.w * v.w;
    }
#pragma unroll
    for (int m = 16; m > 0; m >>= 1) ss += __shfl_xor(ss, m, 32);
    if (lane == 0) {
        int tInB = token & (T_ - 1);
        scores[token] = (tInB == 0) ? 3.402823466e+38f : ss;  // CLS protection
        rnorm[token]  = 1.0f / (sqrtf(ss) + 1e-12f);
    }
}

// ---------------------------------------------------------------------------
// K2: per-batch top-K by bitonic sort in LDS, then sort kept indices ascending.
__global__ __launch_bounds__(1024) void k_topk(const float* __restrict__ scores,
                                               int* __restrict__ keep_idx) {
    __shared__ float sKey[T_];
    __shared__ int   sIdx[T_];
    int b = blockIdx.x, tid = threadIdx.x;
    for (int i = tid; i < T_; i += 1024) { sKey[i] = scores[b * T_ + i]; sIdx[i] = i; }
    __syncthreads();
    // bitonic sort: descending by score, ties -> lower index first
    for (int k = 2; k <= T_; k <<= 1) {
        for (int j = k >> 1; j > 0; j >>= 1) {
            for (int i = tid; i < T_; i += 1024) {
                int ixj = i ^ j;
                if (ixj > i) {
                    float a = sKey[i], c = sKey[ixj];
                    int  ia = sIdx[i], ic = sIdx[ixj];
                    bool aFirst = (a > c) || (a == c && ia < ic);
                    bool desc   = ((i & k) == 0);
                    if (desc ? !aFirst : aFirst) {
                        sKey[i] = c; sKey[ixj] = a;
                        sIdx[i] = ic; sIdx[ixj] = ia;
                    }
                }
            }
            __syncthreads();
        }
    }
    // sort top-K indices ascending (reuse sKey storage as int array)
    int* sK2 = (int*)sKey;
    for (int i = tid; i < K_; i += 1024) sK2[i] = sIdx[i];
    __syncthreads();
    for (int k = 2; k <= K_; k <<= 1) {
        for (int j = k >> 1; j > 0; j >>= 1) {
            for (int i = tid; i < K_; i += 1024) {
                int ixj = i ^ j;
                if (ixj > i) {
                    int a = sK2[i], c = sK2[ixj];
                    bool asc = ((i & k) == 0);
                    if (asc ? (a > c) : (a < c)) { sK2[i] = c; sK2[ixj] = a; }
                }
            }
            __syncthreads();
        }
    }
    for (int i = tid; i < K_; i += 1024) keep_idx[b * K_ + i] = sK2[i];
}

// ---------------------------------------------------------------------------
// K3: gather kept centers as unit-normalized f16 [B,K,C].
__global__ __launch_bounds__(256) void k_gather(const float* __restrict__ x,
                                                const float* __restrict__ rnorm,
                                                const int* __restrict__ keep_idx,
                                                _Float16* __restrict__ keptH) {
    int gk = blockIdx.x;                // b*K + k
    int b  = gk >> 11;                  // / K_
    int t  = keep_idx[gk];
    float rn = rnorm[b * T_ + t];
    const float* src = x + ((size_t)b * T_ + t) * C_;
    _Float16*    dst = keptH + (size_t)gk * C_;
    for (int c = threadIdx.x; c < C_; c += 256)
        dst[c] = (_Float16)(src[c] * rn);
}

// ---------------------------------------------------------------------------
// K4: cosine-similarity matmul (f16 WMMA) + running argmax over K per token.
// Block: 128 threads (4 waves), 16 token rows. Each wave sweeps K/64 n-tiles.
__global__ __launch_bounds__(128) void k_sim_argmax(const float* __restrict__ x,
                                                    const float* __restrict__ rnorm,
                                                    const _Float16* __restrict__ keptH,
                                                    int* __restrict__ assign_idx) {
    __shared__ _Float16 Atile[16][C_];   // 32 KB
    __shared__ float sBestV[4][16];
    __shared__ int   sBestI[4][16];

    int b   = blockIdx.y;
    int t0  = blockIdx.x * 16;
    int tid = threadIdx.x;

    // stage A: 16 rows x 1024 cols, normalized to f16
    {
        const float4* px = (const float4*)(x + ((size_t)b * T_ + t0) * C_);
        for (int i = tid; i < 16 * (C_ / 4); i += 128) {
            int row = i >> 8;            // 256 float4 per row
            int c4  = i & 255;
            float rn = rnorm[b * T_ + t0 + row];
            float4 v = px[(size_t)row * 256 + c4];
            union { uint2 u; _Float16 h[4]; } pk;
            pk.h[0] = (_Float16)(v.x * rn);
            pk.h[1] = (_Float16)(v.y * rn);
            pk.h[2] = (_Float16)(v.z * rn);
            pk.h[3] = (_Float16)(v.w * rn);
            *(uint2*)&Atile[row][c4 * 4] = pk.u;
        }
    }
    __syncthreads();

    int lane  = tid & 31;
    int wave  = tid >> 5;
    int aRow  = lane & 15;               // ISA 16-bit A 16x32 layout
    int aKoff = (lane >> 4) * 8;
    int bKoff = (lane >> 4) * 8;         // B 32x16: same K order as A (8h+s, 16+8h+s)
    int nLane = lane & 15;

    float bestv[8];
    int   besti[8];
#pragma unroll
    for (int r = 0; r < 8; ++r) { bestv[r] = -3.402823466e+38f; besti[r] = 0; }

    for (int j = 0; j < K_ / 64; ++j) {
        int ncol = (j * 4 + wave) * 16 + nLane;
        const _Float16* bp = keptH + ((size_t)b * K_ + ncol) * C_;
        v8f acc = {};
#pragma unroll 4
        for (int c = 0; c < C_; c += 32) {
            union { v16h v; v8h h[2]; } af, bfm;
            af.h[0]  = *(const v8h*)&Atile[aRow][c + aKoff];
            af.h[1]  = *(const v8h*)&Atile[aRow][c + aKoff + 16];
            bfm.h[0] = *(const v8h*)(bp + c + bKoff);
            bfm.h[1] = *(const v8h*)(bp + c + bKoff + 16);
            acc = __builtin_amdgcn_wmma_f32_16x16x32_f16(
                    /*neg_a=*/false, af.v, /*neg_b=*/false, bfm.v,
                    /*c_mod=*/(short)0, acc, /*reuse_a=*/false, /*reuse_b=*/false);
        }
#pragma unroll
        for (int r = 0; r < 8; ++r) {
            float v = acc[r];
            if (v > bestv[r]) { bestv[r] = v; besti[r] = ncol; }
        }
    }
    // cross-lane argmax over the 16 N-columns (lanes sharing a row group)
#pragma unroll
    for (int m = 1; m <= 8; m <<= 1) {
#pragma unroll
        for (int r = 0; r < 8; ++r) {
            float ov = __shfl_xor(bestv[r], m, 32);
            int   oi = __shfl_xor(besti[r], m, 32);
            if (ov > bestv[r] || (ov == bestv[r] && oi < besti[r])) {
                bestv[r] = ov; besti[r] = oi;
            }
        }
    }
    if ((lane & 15) == 0) {
        int base = (lane >> 4) * 8;      // lanes 0..15 -> rows 0..7, 16..31 -> 8..15
#pragma unroll
        for (int r = 0; r < 8; ++r) {
            sBestV[wave][base + r] = bestv[r];
            sBestI[wave][base + r] = besti[r];
        }
    }
    __syncthreads();
    if (tid < 16) {
        float bv = sBestV[0][tid]; int bi = sBestI[0][tid];
        for (int w = 1; w < 4; ++w) {
            float v = sBestV[w][tid]; int i2 = sBestI[w][tid];
            if (v > bv || (v == bv && i2 < bi)) { bv = v; bi = i2; }
        }
        assign_idx[b * T_ + t0 + tid] = bi;
    }
}

// ---------------------------------------------------------------------------
// K5: kept centers self-assign.
__global__ __launch_bounds__(256) void k_selfassign(const int* __restrict__ keep_idx,
                                                    int* __restrict__ assign_idx) {
    int gk = blockIdx.x * 256 + threadIdx.x;   // b*K + k
    if (gk >= B_ * K_) return;
    int b = gk >> 11;
    int k = gk & (K_ - 1);
    assign_idx[b * T_ + keep_idx[gk]] = k;
}

// ---------------------------------------------------------------------------
// K6: segment-sum scatter (atomics into d_out used as sums) + counts.
__global__ __launch_bounds__(256) void k_scatter(const float* __restrict__ x,
                                                 const int* __restrict__ assign_idx,
                                                 float* __restrict__ sums,
                                                 float* __restrict__ cnt) {
    int gt = blockIdx.x;                 // b*T + t
    int b  = gt >> 12;                   // / T_
    int a  = assign_idx[gt];
    size_t sbase = ((size_t)b * K_ + a) * C_;
    const float* src = x + (size_t)gt * C_;
    for (int c = threadIdx.x; c < C_; c += 256)
        atomicAdd(&sums[sbase + c], src[c]);
    if (threadIdx.x == 0) atomicAdd(&cnt[b * K_ + a], 1.0f);
}

// ---------------------------------------------------------------------------
// K7: out = (1-ALPHA)*x_kept + ALPHA * sums/cnt   (in place on d_out).
__global__ __launch_bounds__(256) void k_finalize(const float* __restrict__ x,
                                                  const int* __restrict__ keep_idx,
                                                  const float* __restrict__ cnt,
                                                  float* __restrict__ out) {
    int gk = blockIdx.x;                 // b*K + k
    int b  = gk >> 11;
    int t  = keep_idx[gk];
    float inv = 1.0f / cnt[gk];          // SIZE_DELTA = 0
    const float* src = x + ((size_t)b * T_ + t) * C_;
    float*       o   = out + (size_t)gk * C_;
    for (int c = threadIdx.x; c < C_; c += 256) {
        float s = o[c];
        o[c] = 0.85f * src[c] + 0.15f * s * inv;
    }
}

// ---------------------------------------------------------------------------
extern "C" void kernel_launch(void* const* d_in, const int* in_sizes, int n_in,
                              void* d_out, int out_size, void* d_ws, size_t ws_size,
                              hipStream_t stream) {
    const float* x = (const float*)d_in[0];
    // (layer_idx, requested_r are fixed by setup: K_ = T_ - 2048)

    char* ws = (char*)d_ws;
    float* scores     = (float*)ws;  ws += (size_t)B_ * T_ * 4;   // 128 KB
    float* rnorm      = (float*)ws;  ws += (size_t)B_ * T_ * 4;   // 128 KB
    int*   keep_idx   = (int*)ws;    ws += (size_t)B_ * K_ * 4;   //  64 KB
    int*   assign_idx = (int*)ws;    ws += (size_t)B_ * T_ * 4;   // 128 KB
    float* cnt        = (float*)ws;  ws += (size_t)B_ * K_ * 4;   //  64 KB
    _Float16* keptH   = (_Float16*)ws;                            //  32 MB

    float* out = (float*)d_out;

    hipMemsetAsync(d_out, 0, (size_t)out_size * sizeof(float), stream);
    hipMemsetAsync(cnt, 0, (size_t)B_ * K_ * sizeof(float), stream);

    k_scores    <<<B_ * T_ / 8, 256, 0, stream>>>(x, scores, rnorm);
    k_topk      <<<B_, 1024, 0, stream>>>(scores, keep_idx);
    k_gather    <<<B_ * K_, 256, 0, stream>>>(x, rnorm, keep_idx, keptH);
    k_sim_argmax<<<dim3(T_ / 16, B_), 128, 0, stream>>>(x, rnorm, keptH, assign_idx);
    k_selfassign<<<(B_ * K_) / 256, 256, 0, stream>>>(keep_idx, assign_idx);
    k_scatter   <<<B_ * T_, 256, 0, stream>>>(x, assign_idx, out, cnt);
    k_finalize  <<<B_ * K_, 256, 0, stream>>>(x, keep_idx, cnt, out);
}
